// RING_2585570312533
// MI455X (gfx1250) — compile-verified, matrix operand order checked by reading, not run on hardware
//
#include <hip/hip_runtime.h>

typedef __attribute__((ext_vector_type(16))) _Float16 v16h;
typedef __attribute__((ext_vector_type(8)))  _Float16 v8h;
typedef __attribute__((ext_vector_type(2)))  _Float16 h2;
typedef __attribute__((ext_vector_type(8)))  float    v8f;
typedef __attribute__((ext_vector_type(4)))  unsigned int u32x4;
typedef __attribute__((ext_vector_type(8)))  int      i32x8;
typedef __attribute__((ext_vector_type(4)))  int      i32x4;

#define NPIX 120      // image side, s, freq, angles (all 120 here)
#define NANG 120
#define NPAD 128      // padded GEMM dimension
#define TWO_PI 6.28318530717958647692f

__global__ __launch_bounds__(256)
void radon_dft_norm_kernel(const float* __restrict__ bev, float* __restrict__ out)
{
    // ---- LDS: ~91.4 KB total (fits 3 blocks per 320KB WGP) ----
    __shared__ __align__(16) float    s_img[NPIX * NPIX];   // 57600 B  source image (TDM target)
    __shared__ __align__(16) _Float16 s_sino[NPAD * NPAD];  // 32768 B  sinogram [angle][s], zero padded
    __shared__ h2    s_cs[NPIX];                            // packed (cos,sin)(2*pi*m/120) as f16
    __shared__ float s_xsum[NPAD];                          // per-frequency angle-sum of spectra
    __shared__ float s_scale;

    const int tid = threadIdx.x;
    const int b   = blockIdx.x;
    const float c = (NPIX - 1) * 0.5f;          // 59.5

    // ---- TDM: DMA the 120x120 f32 image global -> LDS (issued once, by wave 0) ----
    if (tid == 0) {
        const unsigned long long gaddr = (unsigned long long)(bev + (size_t)b * (NPIX * NPIX));
        const unsigned lds_base = (unsigned)(unsigned long long)(const void*)&s_img[0]; // addr[31:0] = LDS offset
        u32x4 g0;
        g0[0] = 1u;                                   // count=1, user descriptor
        g0[1] = lds_base;                             // D#.lds_addr
        g0[2] = (unsigned)gaddr;                      // D#.global_addr[31:0]
        g0[3] = (unsigned)((gaddr >> 32) & 0x01FFFFFFull) | (2u << 30);  // addr[56:32] | type=2
        i32x8 g1;
        g1[0] = (int)(2u << 16);                      // workgroup_mask=0, data_size=2 (4 bytes)
        g1[1] = (int)((NPIX * NPIX & 0xFFFF) << 16);  // tensor_dim0[15:0] @ bits 63:48
        g1[2] = (int)((NPIX * NPIX) >> 16) | (1 << 16); // tensor_dim0[31:16] | tensor_dim1=1
        g1[3] = (int)((NPIX * NPIX) << 16);           // tile_dim0 = 14400 @ bits 127:112
        g1[4] = 1;                                    // tile_dim1 = 1, tile_dim2 = 0
        g1[5] = NPIX * NPIX;                          // tensor_dim0_stride[31:0]
        g1[6] = 0;                                    // stride hi, tensor_dim1_stride lo
        g1[7] = 0;
        i32x4 gz4 = {0, 0, 0, 0};                     // groups 2/3 unused (<=2D tensor)
        i32x8 gz8 = {0, 0, 0, 0, 0, 0, 0, 0};
        // amdgpu-toolchain (clang-23) 6-arg form: (g0, g1, g2, g3, g_extra, cpol)
        __builtin_amdgcn_tensor_load_to_lds(g0, g1, gz4, gz4, gz8, 0);
    }

    // ---- init padded sinogram / accumulators / packed f16 twiddle table ----
    for (int i = tid; i < NPAD * NPAD; i += 256) s_sino[i] = (_Float16)0.0f;
    for (int i = tid; i < NPAD; i += 256)        s_xsum[i] = 0.0f;
    if (tid < NPIX) {
        float sv, cv;
        __sincosf(TWO_PI * (float)tid / (float)NPIX, &sv, &cv);
        h2 p; p.x = (_Float16)cv; p.y = (_Float16)sv;
        s_cs[tid] = p;
    }
    __builtin_amdgcn_s_wait_tensorcnt(0);   // wave 0: image DMA complete; others: free
    __syncthreads();

    // ---- Radon: each thread owns (angle, s-row) pairs, sums 120 bilinear taps ----
    const float dth = TWO_PI / (float)(NANG - 1);   // np.linspace(0, 2pi, 120)
    for (int p = tid; p < NANG * NPIX; p += 256) {
        const int a    = p / NPIX;
        const int srow = p - a * NPIX;
        float st, ct;
        __sincosf((float)a * dth, &st, &ct);
        const float sv = (float)srow - c;
        float px = c + sv * ct + c * st;            // t index 0 -> t_val = -c
        float py = c + sv * st - c * ct;
        float acc = 0.0f;
        for (int t = 0; t < NPIX; ++t) {
            const float x0f = floorf(px), y0f = floorf(py);
            const float wx1 = px - x0f,   wy1 = py - y0f;
            const float wx0 = 1.0f - wx1, wy0 = 1.0f - wy1;
            const int x0 = (int)x0f, y0 = (int)y0f;
            const int x1 = x0 + 1,   y1 = y0 + 1;
            // clip + validity mask (matches reference: value * (w * valid))
            const float mx0 = (x0 >= 0 && x0 < NPIX) ? 1.0f : 0.0f;
            const float mx1 = (x1 >= 0 && x1 < NPIX) ? 1.0f : 0.0f;
            const float my0 = (y0 >= 0 && y0 < NPIX) ? 1.0f : 0.0f;
            const float my1 = (y1 >= 0 && y1 < NPIX) ? 1.0f : 0.0f;
            const int xc0 = min(max(x0, 0), NPIX - 1);
            const int xc1 = min(max(x1, 0), NPIX - 1);
            const int yc0 = min(max(y0, 0), NPIX - 1);
            const int yc1 = min(max(y1, 0), NPIX - 1);
            const float v00 = s_img[yc0 * NPIX + xc0];
            const float v01 = s_img[yc0 * NPIX + xc1];
            const float v10 = s_img[yc1 * NPIX + xc0];
            const float v11 = s_img[yc1 * NPIX + xc1];
            acc += wy0 * my0 * (wx0 * mx0 * v00 + wx1 * mx1 * v01)
                 + wy1 * my1 * (wx0 * mx0 * v10 + wx1 * mx1 * v11);
            px -= st;   // d(px)/dt = -sin(theta)
            py += ct;   // d(py)/dt = +cos(theta)
        }
        s_sino[a * NPAD + srow] = (_Float16)acc;
    }
    __syncthreads();

    // ---- DFT magnitude via WMMA: D = sino(128x128,f16) x {cos,sin}(128x128,f16) ----
    // wave w owns M-tile (angles 16w..16w+15); loops 8 N-tiles x 4 K-steps of 32.
    const int lane = tid & 31;
    const int wv   = tid >> 5;        // 0..7  -> m-tile
    const int half = lane >> 4;       // 0/1 lane half
    const int lrow = lane & 15;       // A: m within tile, B/D: n within tile
    const int mrow = wv * 16 + lrow;  // A-matrix row this lane feeds
    const _Float16* arow = &s_sino[mrow * NPAD];

    for (int nt = 0; nt < 8; ++nt) {
        v8f acc_re = {};
        v8f acc_im = {};
        const int ncol = nt * 16 + lrow;
        for (int kk = 0; kk < 4; ++kk) {
            const int k0 = kk * 32;
            // ISA 7.12.2 16-bit A 16x32 layout: lane needs two contiguous 8-half runs
            // -> two ds_load_b128 instead of 16 scalar u16 loads.
            const v8h alo = *(const v8h*)(arow + k0 + 8 * half);
            const v8h ahi = *(const v8h*)(arow + k0 + 16 + 8 * half);
            const v16h af = __builtin_shufflevector(alo, ahi,
                                0, 1, 2, 3, 4, 5, 6, 7, 8, 9, 10, 11, 12, 13, 14, 15);
            v16h bc, bs;
            #pragma unroll
            for (int v = 0; v < 8; ++v) {
                const int kloc = ((v < 4) ? 2 * v : 16 + 2 * (v - 4)) + 8 * half;
                const int kg   = k0 + kloc;
                // DFT twiddles: angle index (k*n) mod 120; one packed b32 gather each
                const h2 t0 = s_cs[(kg * ncol) % NANG];
                const h2 t1 = s_cs[((kg + 1) * ncol) % NANG];
                bc[2 * v]     = t0.x;
                bc[2 * v + 1] = t1.x;
                bs[2 * v]     = t0.y;
                bs[2 * v + 1] = t1.y;
            }
            acc_re = __builtin_amdgcn_wmma_f32_16x16x32_f16(
                false, af, false, bc, (short)0, acc_re, false, false);
            acc_im = __builtin_amdgcn_wmma_f32_16x16x32_f16(
                false, af, false, bs, (short)0, acc_im, false, false);
        }
        // spec = sqrt(re^2 + im^2 + eps), masked for zero-padded angles >= 120,
        // reduced over the 8 m-values this lane holds, then LDS-atomic into xsum[n].
        float part = 0.0f;
        #pragma unroll
        for (int v = 0; v < 8; ++v) {
            const int mg = wv * 16 + v + 8 * half;   // C/D layout: m = vgpr + 8*half
            const float re = acc_re[v];
            const float im = acc_im[v];
            const float sp = __builtin_amdgcn_sqrtf(re * re + im * im + 1e-15f);
            part += (mg < NANG) ? sp : 0.0f;
        }
        atomicAdd(&s_xsum[nt * 16 + lrow], part);    // lanes L and L^16 share n -> ds_add_f32
    }
    __syncthreads();

    // ---- L2 normalize the 120-vector (global scale of DFT cancels here) ----
    if (tid == 0) {
        float ss = 0.0f;
        for (int i = 0; i < NANG; ++i) { const float v = s_xsum[i]; ss += v * v; }
        s_scale = 1.0f / fmaxf(__builtin_amdgcn_sqrtf(ss), 1e-12f);
    }
    __syncthreads();
    if (tid < NANG) out[(size_t)b * NANG + tid] = s_xsum[tid] * s_scale;
}

extern "C" void kernel_launch(void* const* d_in, const int* in_sizes, int n_in,
                              void* d_out, int out_size, void* d_ws, size_t ws_size,
                              hipStream_t stream) {
    (void)n_in; (void)out_size; (void)d_ws; (void)ws_size;
    const float* bev = (const float*)d_in[0];
    float* out = (float*)d_out;
    const int B = in_sizes[0] / (NPIX * NPIX);   // 1024
    hipLaunchKernelGGL(radon_dft_norm_kernel, dim3(B), dim3(256), 0, stream, bev, out);
}
